// SpectralMultiheadAttention_76785425318191
// MI455X (gfx1250) — matrix operand order repaired and checked by reading, not compiled, not run on hardware
//
#include <hip/hip_runtime.h>
#include <hip/hip_bf16.h>

typedef unsigned short u16;
typedef __attribute__((ext_vector_type(16))) __bf16 v16bf;
typedef __attribute__((ext_vector_type(8)))  float  v8f;
typedef __attribute__((ext_vector_type(8)))  int    v8i;

static constexpr int S = 2048;
static constexpr int D = 1024;
static constexpr int H = 16;
static constexpr int HD = 64;

// ---------- helpers ----------
__device__ __forceinline__ u16 f2bf(float f) {
    unsigned u = __builtin_bit_cast(unsigned, f);
    u += 0x7FFFu + ((u >> 16) & 1u);          // round to nearest even
    return (u16)(u >> 16);
}

__device__ __forceinline__ v8f v8f_zero() {
    v8f z;
#pragma unroll
    for (int i = 0; i < 8; ++i) z[i] = 0.0f;
    return z;
}

// Negate a bf16x16 fragment by flipping sign bits (used to get  -A  operand).
__device__ __forceinline__ v16bf negbf(v16bf x) {
    v8i u = __builtin_bit_cast(v8i, x);
#pragma unroll
    for (int i = 0; i < 8; ++i) u[i] ^= (int)0x80008000u;
    return __builtin_bit_cast(v16bf, u);
}

// B-fragment (32x16 bf16, lane = N column ln; element s = K 8*hi+s (s<8),
// 16+8*hi+(s-8) (s>=8)): strided gather, p points at (K base row, column), ld = row stride.
__device__ __forceinline__ v16bf ldB(const u16* p, int hi, int ld) {
    v16bf v;
#pragma unroll
    for (int s = 0; s < 16; ++s) {
        const int k = (s < 8) ? (8 * hi + s) : (16 + 8 * hi + (s - 8));
        v[s] = __builtin_bit_cast(__bf16, p[(size_t)k * ld]);
    }
    return v;
}

// A-fragment (16x32 bf16, lane = M row; lanes 0-15 hold K 0-7 & 16-23,
// lanes 16-31 hold K 8-15 & 24-31): two 16-byte loads.
struct AFrag32B { uint4 lo; uint4 hi; };
__device__ __forceinline__ v16bf ldA(const u16* rowk0, int hi) {
    AFrag32B u;
    u.lo = *reinterpret_cast<const uint4*>(rowk0 + 8 * hi);
    u.hi = *reinterpret_cast<const uint4*>(rowk0 + 16 + 8 * hi);
    return __builtin_bit_cast(v16bf, u);
}

__device__ __forceinline__ v8f wmma_bf16(v16bf a, v16bf b, v8f c) {
    return __builtin_amdgcn_wmma_f32_16x16x32_bf16(false, a, false, b,
                                                   (short)0, c, false, false);
}

// ---------- stage 0: fp32 -> bf16 convert / transpose-convert ----------
__global__ void cvt_kernel(const float* __restrict__ src, u16* __restrict__ dst, int n) {
    int i = blockIdx.x * blockDim.x + threadIdx.x;
    if (i < n) dst[i] = f2bf(src[i]);
}

// Wt[d][e] = bf16(W[e][d])  (D x D)
__global__ void cvt_tr_kernel(const float* __restrict__ W, u16* __restrict__ Wt) {
    int e = blockIdx.x * 16 + threadIdx.x;
    int d = blockIdx.y * 16 + threadIdx.y;
    Wt[(size_t)d * D + e] = f2bf(W[(size_t)e * D + d]);
}

// ---------- stage 1: complex projection  out = x @ W^T + b ----------
// mode 0: q (scaled by 1/8, row-major)   mode 1: k (per-feature transposed (D,S))
// mode 2: v (row-major)
__global__ void __launch_bounds__(128) proj_kernel(
    const u16* __restrict__ xr, const u16* __restrict__ xi,
    const u16* __restrict__ wtr, const u16* __restrict__ wti,
    const float* __restrict__ br, const float* __restrict__ bi,
    u16* __restrict__ outr, u16* __restrict__ outi, int mode) {
    const int lane = threadIdx.x & 31, wave = threadIdx.x >> 5;
    const int ln = lane & 15, hi = lane >> 4;
    const int s0 = (blockIdx.x * 4 + wave) * 16;
    const int n0 = blockIdx.y * 32;

    v8f accR[2], accI[2];
#pragma unroll
    for (int nt = 0; nt < 2; ++nt) { accR[nt] = v8f_zero(); accI[nt] = v8f_zero(); }

    const u16* xrow_r = xr + (size_t)(s0 + ln) * D;
    const u16* xrow_i = xi + (size_t)(s0 + ln) * D;

    for (int k0 = 0; k0 < D; k0 += 32) {
        v16bf axr  = ldA(xrow_r + k0, hi);
        v16bf axi  = ldA(xrow_i + k0, hi);
        v16bf axin = negbf(axi);
        const int krow = k0;
#pragma unroll
        for (int nt = 0; nt < 2; ++nt) {
            v16bf bwr = ldB(wtr + (size_t)krow * D + n0 + nt * 16 + ln, hi, D);
            v16bf bwi = ldB(wti + (size_t)krow * D + n0 + nt * 16 + ln, hi, D);
            accR[nt] = wmma_bf16(axr,  bwr, accR[nt]);   // + xr*Wr
            accR[nt] = wmma_bf16(axin, bwi, accR[nt]);   // - xi*Wi
            accI[nt] = wmma_bf16(axr,  bwi, accI[nt]);   // + xr*Wi
            accI[nt] = wmma_bf16(axi,  bwr, accI[nt]);   // + xi*Wr
        }
    }

#pragma unroll
    for (int nt = 0; nt < 2; ++nt) {
        const int n = n0 + nt * 16 + ln;
        const float vbr = br[n], vbi = bi[n];
#pragma unroll
        for (int g = 0; g < 8; ++g) {
            const int row = s0 + g + 8 * hi;
            float orv = accR[nt][g] + vbr;
            float oiv = accI[nt][g] + vbi;
            if (mode == 0) { orv *= 0.125f; oiv *= 0.125f; }   // fold 1/sqrt(HD)
            if (mode == 1) {                                   // K transposed (D,S)
                outr[(size_t)n * S + row] = f2bf(orv);
                outi[(size_t)n * S + row] = f2bf(oiv);
            } else {
                outr[(size_t)row * D + n] = f2bf(orv);
                outi[(size_t)row * D + n] = f2bf(oiv);
            }
        }
    }
}

// ---------- stage 2: complex flash attention (one 16-row Q block / wave) ----------
__global__ void __launch_bounds__(128) attn_kernel(
    const u16* __restrict__ qr, const u16* __restrict__ qi,
    const u16* __restrict__ ktr, const u16* __restrict__ kti,
    const u16* __restrict__ vr, const u16* __restrict__ vi,
    u16* __restrict__ resr, u16* __restrict__ resi) {
    __shared__ alignas(32) u16 ldsP[4][2][16 * 32];   // per-wave probs tiles (r,i)

    const int lane = threadIdx.x & 31, wave = threadIdx.x >> 5;
    const int ln = lane & 15, hi = lane >> 4;
    const int h = blockIdx.y;
    const int s0 = (blockIdx.x * 4 + wave) * 16;
    const int kkrow = ln + 16 * hi;

    // Q fragments (row = s0+ln, two K chunks of 32 over HD=64); Q pre-scaled by 1/8.
    const u16* qrow_r = qr + (size_t)(s0 + ln) * D + h * HD;
    const u16* qrow_i = qi + (size_t)(s0 + ln) * D + h * HD;
    v16bf aqr[2], aqi[2], aqrn[2];
#pragma unroll
    for (int c = 0; c < 2; ++c) {
        aqr[c]  = ldA(qrow_r + c * 32, hi);
        aqi[c]  = ldA(qrow_i + c * 32, hi);
        aqrn[c] = negbf(aqr[c]);
    }

    float rm[8], Zr[8], Zi[8];
    v8f accR[4], accI[4];
#pragma unroll
    for (int g = 0; g < 8; ++g) { rm[g] = -3.0e38f; Zr[g] = 0.f; Zi[g] = 0.f; }
#pragma unroll
    for (int j = 0; j < 4; ++j) { accR[j] = v8f_zero(); accI[j] = v8f_zero(); }

    u16* myPr = &ldsP[wave][0][0];
    u16* myPi = &ldsP[wave][1][0];

    for (int T = 0; T < S; T += 32) {
        // --- scores: two 16x16 complex tiles;  s = q * conj(k) ---
        v8f sr[2], si[2];
#pragma unroll
        for (int t = 0; t < 2; ++t) {
            sr[t] = v8f_zero(); si[t] = v8f_zero();
            const int t0 = T + 16 * t;
#pragma unroll
            for (int c = 0; c < 2; ++c) {
                v16bf bkr = ldB(ktr + (size_t)(h * HD + c * 32) * S + t0 + ln, hi, S);
                v16bf bki = ldB(kti + (size_t)(h * HD + c * 32) * S + t0 + ln, hi, S);
                sr[t] = wmma_bf16(aqr[c],  bkr, sr[t]);   // + qr*kr
                sr[t] = wmma_bf16(aqi[c],  bki, sr[t]);   // + qi*ki
                si[t] = wmma_bf16(aqi[c],  bkr, si[t]);   // + qi*kr
                si[t] = wmma_bf16(aqrn[c], bki, si[t]);   // - qr*ki
            }
        }

        // --- streaming complex softmax (max over real part, per row) ---
#pragma unroll
        for (int g = 0; g < 8; ++g) {
            float tmax = fmaxf(sr[0][g], sr[1][g]);
#pragma unroll
            for (int m = 1; m < 16; m <<= 1) tmax = fmaxf(tmax, __shfl_xor(tmax, m, 32));
            const float nm = fmaxf(rm[g], tmax);
            const float sc = __expf(rm[g] - nm);
            rm[g] = nm;
            Zr[g] *= sc; Zi[g] *= sc;
#pragma unroll
            for (int j = 0; j < 4; ++j) { accR[j][g] *= sc; accI[j][g] *= sc; }

            const int row = g + 8 * hi;
#pragma unroll
            for (int t = 0; t < 2; ++t) {
                const float e = __expf(sr[t][g] - nm);
                float sv, cv;
                __sincosf(si[t][g], &sv, &cv);
                const float pr = e * cv, pi = e * sv;     // complex exp
                Zr[g] += pr; Zi[g] += pi;
                myPr[row * 32 + 16 * t + ln] = f2bf(pr);
                myPi[row * 32 + 16 * t + ln] = f2bf(pi);
            }
        }

        // same-wave LDS store->load re-layout (C layout -> A layout)
        asm volatile("s_wait_dscnt 0x0" ::: "memory");
        v16bf apr  = ldA(myPr + ln * 32, hi);
        v16bf api  = ldA(myPi + ln * 32, hi);
        v16bf apin = negbf(api);

        // --- accumulate numerator:  num += P * V  (complex) ---
        const u16* vr_p = vr + (size_t)T * D + h * HD;
        const u16* vi_p = vi + (size_t)T * D + h * HD;
#pragma unroll
        for (int j = 0; j < 4; ++j) {
            v16bf bvr = ldB(vr_p + 16 * j + ln, hi, D);
            v16bf bvi = ldB(vi_p + 16 * j + ln, hi, D);
            accR[j] = wmma_bf16(apr,  bvr, accR[j]);
            accR[j] = wmma_bf16(apin, bvi, accR[j]);
            accI[j] = wmma_bf16(apr,  bvi, accI[j]);
            accI[j] = wmma_bf16(api,  bvr, accI[j]);
        }
    }

    // reduce complex Z across the 16 columns
#pragma unroll
    for (int g = 0; g < 8; ++g) {
        float zr = Zr[g], zi = Zi[g];
#pragma unroll
        for (int m = 1; m < 16; m <<= 1) { zr += __shfl_xor(zr, m, 32); zi += __shfl_xor(zi, m, 32); }
        Zr[g] = zr; Zi[g] = zi;
    }

    // res = num * conj(Z) / |Z|^2
#pragma unroll
    for (int j = 0; j < 4; ++j) {
#pragma unroll
        for (int g = 0; g < 8; ++g) {
            const float zr = Zr[g], zi = Zi[g];
            const float inv = 1.0f / (zr * zr + zi * zi);
            const float nr = accR[j][g], ni = accI[j][g];
            const int row = s0 + g + 8 * hi;
            const int col = h * HD + j * 16 + ln;
            resr[(size_t)row * D + col] = f2bf((nr * zr + ni * zi) * inv);
            resi[(size_t)row * D + col] = f2bf((ni * zr - nr * zi) * inv);
        }
    }
}

// ---------- stage 3: output projection -> interleaved complex64 ----------
__global__ void __launch_bounds__(128) outproj_kernel(
    const u16* __restrict__ xr, const u16* __restrict__ xi,
    const u16* __restrict__ wtr, const u16* __restrict__ wti,
    const float* __restrict__ br, const float* __restrict__ bi,
    float* __restrict__ out) {
    const int lane = threadIdx.x & 31, wave = threadIdx.x >> 5;
    const int ln = lane & 15, hi = lane >> 4;
    const int s0 = (blockIdx.x * 4 + wave) * 16;
    const int n0 = blockIdx.y * 32;

    v8f accR[2], accI[2];
#pragma unroll
    for (int nt = 0; nt < 2; ++nt) { accR[nt] = v8f_zero(); accI[nt] = v8f_zero(); }

    const u16* xrow_r = xr + (size_t)(s0 + ln) * D;
    const u16* xrow_i = xi + (size_t)(s0 + ln) * D;

    for (int k0 = 0; k0 < D; k0 += 32) {
        v16bf axr  = ldA(xrow_r + k0, hi);
        v16bf axi  = ldA(xrow_i + k0, hi);
        v16bf axin = negbf(axi);
        const int krow = k0;
#pragma unroll
        for (int nt = 0; nt < 2; ++nt) {
            v16bf bwr = ldB(wtr + (size_t)krow * D + n0 + nt * 16 + ln, hi, D);
            v16bf bwi = ldB(wti + (size_t)krow * D + n0 + nt * 16 + ln, hi, D);
            accR[nt] = wmma_bf16(axr,  bwr, accR[nt]);
            accR[nt] = wmma_bf16(axin, bwi, accR[nt]);
            accI[nt] = wmma_bf16(axr,  bwi, accI[nt]);
            accI[nt] = wmma_bf16(axi,  bwr, accI[nt]);
        }
    }

    float2* o = reinterpret_cast<float2*>(out);
#pragma unroll
    for (int nt = 0; nt < 2; ++nt) {
        const int n = n0 + nt * 16 + ln;
        const float vbr = br[n], vbi = bi[n];
#pragma unroll
        for (int g = 0; g < 8; ++g) {
            const int row = s0 + g + 8 * hi;
            float2 c;
            c.x = accR[nt][g] + vbr;
            c.y = accI[nt][g] + vbi;
            o[(size_t)row * D + n] = c;
        }
    }
}

// ---------- host ----------
extern "C" void kernel_launch(void* const* d_in, const int* in_sizes, int n_in,
                              void* d_out, int out_size, void* d_ws, size_t ws_size,
                              hipStream_t stream) {
    (void)in_sizes; (void)n_in; (void)out_size; (void)ws_size;
    const float* x_r  = (const float*)d_in[0];
    const float* x_i  = (const float*)d_in[1];
    const float* W_r[4] = { (const float*)d_in[2], (const float*)d_in[6],
                            (const float*)d_in[10], (const float*)d_in[14] };
    const float* W_i[4] = { (const float*)d_in[3], (const float*)d_in[7],
                            (const float*)d_in[11], (const float*)d_in[15] };
    const float* b_r[4] = { (const float*)d_in[4], (const float*)d_in[8],
                            (const float*)d_in[12], (const float*)d_in[16] };
    const float* b_i[4] = { (const float*)d_in[5], (const float*)d_in[9],
                            (const float*)d_in[13], (const float*)d_in[17] };

    const size_t SD = (size_t)S * D, DD = (size_t)D * D;
    u16* p = (u16*)d_ws;
    u16* xb_r = p;            p += SD;
    u16* xb_i = p;            p += SD;
    u16* wt_r[4]; u16* wt_i[4];
    for (int w = 0; w < 4; ++w) { wt_r[w] = p; p += DD; wt_i[w] = p; p += DD; }
    u16* qb_r  = p; p += SD;  u16* qb_i  = p; p += SD;
    u16* ktb_r = p; p += SD;  u16* ktb_i = p; p += SD;
    u16* vb_r  = p; p += SD;  u16* vb_i  = p; p += SD;
    u16* rs_r  = p; p += SD;  u16* rs_i  = p; p += SD;

    // stage 0: converts
    cvt_kernel<<<(int)(SD / 256), 256, 0, stream>>>(x_r, xb_r, (int)SD);
    cvt_kernel<<<(int)(SD / 256), 256, 0, stream>>>(x_i, xb_i, (int)SD);
    dim3 trb(16, 16), trg(D / 16, D / 16);
    for (int w = 0; w < 4; ++w) {
        cvt_tr_kernel<<<trg, trb, 0, stream>>>(W_r[w], wt_r[w]);
        cvt_tr_kernel<<<trg, trb, 0, stream>>>(W_i[w], wt_i[w]);
    }

    // stage 1: Q (scaled), K (transposed), V
    dim3 pg(S / 64, D / 32);
    proj_kernel<<<pg, 128, 0, stream>>>(xb_r, xb_i, wt_r[0], wt_i[0], b_r[0], b_i[0],
                                        qb_r, qb_i, 0);
    proj_kernel<<<pg, 128, 0, stream>>>(xb_r, xb_i, wt_r[1], wt_i[1], b_r[1], b_i[1],
                                        ktb_r, ktb_i, 1);
    proj_kernel<<<pg, 128, 0, stream>>>(xb_r, xb_i, wt_r[2], wt_i[2], b_r[2], b_i[2],
                                        vb_r, vb_i, 2);

    // stage 2: complex flash attention
    dim3 ag(S / 64, H);
    attn_kernel<<<ag, 128, 0, stream>>>(qb_r, qb_i, ktb_r, ktb_i, vb_r, vb_i, rs_r, rs_i);

    // stage 3: output projection -> complex64 interleaved
    outproj_kernel<<<pg, 128, 0, stream>>>(rs_r, rs_i, wt_r[3], wt_i[3], b_r[3], b_i[3],
                                           (float*)d_out);
}
